// MJOLNIR_R_79517024518893
// MI455X (gfx1250) — matrix operand order repaired and checked by reading, not compiled, not run on hardware
//
#include <hip/hip_runtime.h>
#include <hip/hip_bf16.h>

// ---------------------------------------------------------------------------
// Types for CDNA5 WMMA
// ---------------------------------------------------------------------------
typedef __attribute__((ext_vector_type(16))) __bf16 v16bf;
typedef __attribute__((ext_vector_type(8)))  float  v8f;

union Frag {
    uint4  q[2];   // 32 bytes = 8 VGPRs
    v16bf  v;
};

// bf16 helpers -- use the native hardware conversion (single v_cvt, RNE)
__device__ __forceinline__ unsigned f2bf(float f) {
    union { __bf16 h; unsigned short s; } c;
    c.h = (__bf16)f;
    return (unsigned)c.s;
}
__device__ __forceinline__ float bf2f(unsigned short h) {
    union { unsigned short s; __bf16 h; } c;
    c.s = h;
    return (float)c.h;
}

// ---------------------------------------------------------------------------
// bf16 WMMA GEMM with K-contiguous B:
//     C[m][n] (or Ct[n][m] if TRANSC) = act( sum_k A[m][k] * Bt[n][k] )
//
// CONTRACT (no edge handling -- caller guarantees):
//   - Kact % 32 == 0; M % 128 == 0
//   - A row-major [M][Kact] (fp32 if AF32 else bf16), leading dim lda (%4/%8)
//   - Bt bf16 [N][Kact], leading dim ldbt % 8 == 0; rows n0..n0+127 readable
//     for every launched N-tile (pad rows are zero)
//   - C: TRANSC=0 -> bf16 [M][*] ld ldc, cols < Nstore written
//        TRANSC=1 -> bf16 [N][M] ld ldc (ldc == M), rows(n) < Nstore written
//
// Block tile 128x128, 8 waves (4 M x 2 N), wave tile 32x64 (2x4 WMMA frags),
// K stepped by 32, double-buffered LDS, register-staged software pipeline:
//   load(k+1) -> WMMA(k) -> LDS-store(k+1) -> barrier
// Staged data lives in named scalar uint4/float4 (no local arrays -> no spill).
// ---------------------------------------------------------------------------
#define SK 56   // LDS row stride in bf16 elements (112B: 16B-aligned, conflict-free)

template<bool AF32, bool RELU, bool TRANSC>
__global__ __launch_bounds__(256, 1)
void gemm_bf16_wmma(const void* __restrict__ Ap,
                    const unsigned short* __restrict__ Bt,
                    unsigned short* __restrict__ Cp,
                    int Kact, int lda, int ldbt, int ldc, int Nstore)
{
    __shared__ unsigned short lsA[2][128 * SK];   // [m][k]
    __shared__ unsigned short lsB[2][128 * SK];   // [n][k]

    const int tid  = threadIdx.x;
    const int lane = tid & 31;
    const int wave = tid >> 5;
    const size_t m0 = (size_t)blockIdx.y * 128;
    const int    n0 = blockIdx.x * 128;
    const int wm = (wave & 3) * 32;   // wave row offset within block tile
    const int wn = (wave >> 2) * 64;  // wave col offset within block tile
    const int kTiles = Kact >> 5;

    v8f acc[2][4];
    #pragma unroll
    for (int i = 0; i < 2; i++)
        #pragma unroll
        for (int j = 0; j < 4; j++)
            #pragma unroll
            for (int g = 0; g < 8; g++) acc[i][j][g] = 0.0f;

    // ---- register staging state: named scalars only (force VGPRs) ------
    float4 sF0, sF1, sF2, sF3;   // AF32 path
    uint4  sA0, sA1;             // bf16-A path
    uint4  sB0, sB1;             // B path

    // per-thread staging coordinates
    const int arF = tid >> 3;            // f32 A: row within 32-row pass
    const int akF = (tid & 7) * 4;       // f32 A: k offset
    const int arB = tid >> 2;            // bf16 A: row within 64-row pass
    const int akB = (tid & 3) * 8;       // bf16 A: k offset
    const int bNr = tid >> 2;            // B: n row within 64-row pass
    const int bKc = (tid & 3) * 8;       // B: k offset

    auto loadStage = [&](int k0) {
        if constexpr (AF32) {
            const float* Af = (const float*)Ap;
            const float* s0 = Af + (m0 + (size_t)(arF      )) * (size_t)lda + (k0 + akF);
            const float* s1 = Af + (m0 + (size_t)(32  + arF)) * (size_t)lda + (k0 + akF);
            const float* s2 = Af + (m0 + (size_t)(64  + arF)) * (size_t)lda + (k0 + akF);
            const float* s3 = Af + (m0 + (size_t)(96  + arF)) * (size_t)lda + (k0 + akF);
            sF0 = *(const float4*)s0;  __builtin_prefetch((const void*)(s0 + 64), 0, 0);
            sF1 = *(const float4*)s1;  __builtin_prefetch((const void*)(s1 + 64), 0, 0);
            sF2 = *(const float4*)s2;  __builtin_prefetch((const void*)(s2 + 64), 0, 0);
            sF3 = *(const float4*)s3;  __builtin_prefetch((const void*)(s3 + 64), 0, 0);
        } else {
            const unsigned short* Ab = (const unsigned short*)Ap;
            sA0 = *(const uint4*)(Ab + (m0 + (size_t)(arB     )) * (size_t)lda + (k0 + akB));
            sA1 = *(const uint4*)(Ab + (m0 + (size_t)(64 + arB)) * (size_t)lda + (k0 + akB));
        }
        sB0 = *(const uint4*)(Bt + (size_t)(n0 + bNr     ) * (size_t)ldbt + (k0 + bKc));
        sB1 = *(const uint4*)(Bt + (size_t)(n0 + 64 + bNr) * (size_t)ldbt + (k0 + bKc));
    };

    auto storeStage = [&](int buf) {
        if constexpr (AF32) {
            uint2 u;
            u.x = f2bf(sF0.x) | (f2bf(sF0.y) << 16);
            u.y = f2bf(sF0.z) | (f2bf(sF0.w) << 16);
            *(uint2*)&lsA[buf][(arF      ) * SK + akF] = u;
            u.x = f2bf(sF1.x) | (f2bf(sF1.y) << 16);
            u.y = f2bf(sF1.z) | (f2bf(sF1.w) << 16);
            *(uint2*)&lsA[buf][(32  + arF) * SK + akF] = u;
            u.x = f2bf(sF2.x) | (f2bf(sF2.y) << 16);
            u.y = f2bf(sF2.z) | (f2bf(sF2.w) << 16);
            *(uint2*)&lsA[buf][(64  + arF) * SK + akF] = u;
            u.x = f2bf(sF3.x) | (f2bf(sF3.y) << 16);
            u.y = f2bf(sF3.z) | (f2bf(sF3.w) << 16);
            *(uint2*)&lsA[buf][(96  + arF) * SK + akF] = u;
        } else {
            *(uint4*)&lsA[buf][(arB     ) * SK + akB] = sA0;   // 16B aligned (112|16)
            *(uint4*)&lsA[buf][(64 + arB) * SK + akB] = sA1;
        }
        *(uint4*)&lsB[buf][(bNr     ) * SK + bKc] = sB0;
        *(uint4*)&lsB[buf][(64 + bNr) * SK + bKc] = sB1;
    };

    loadStage(0);
    storeStage(0);
    __syncthreads();

    const int half = lane >> 4;   // lane half selects K sub-runs
    const int r16  = lane & 15;

    for (int kt = 0; kt < kTiles; kt++) {
        const int  buf  = kt & 1;
        const bool more = (kt + 1 < kTiles);
        if (more) loadStage((kt + 1) * 32);   // global loads fly under the WMMAs

        // A fragment (16x32 bf16): lane<16 -> K[0..7],[16..23]; lane>=16 -> K[8..15],[24..31]
        Frag a[2];
        #pragma unroll
        for (int mi = 0; mi < 2; mi++) {
            const unsigned short* base = &lsA[buf][(wm + mi * 16 + r16) * SK + half * 8];
            a[mi].q[0] = *(const uint4*)base;
            a[mi].q[1] = *(const uint4*)(base + 16);
        }
        // B fragment (32x16 bf16): lane N=lane%16; lane<16 -> K[0..7],[16..23]; lane>=16 -> K[8..15],[24..31]
        Frag b[4];
        #pragma unroll
        for (int ni = 0; ni < 4; ni++) {
            const unsigned short* base = &lsB[buf][(wn + ni * 16 + r16) * SK + half * 8];
            b[ni].q[0] = *(const uint4*)base;
            b[ni].q[1] = *(const uint4*)(base + 16);
        }
        #pragma unroll
        for (int mi = 0; mi < 2; mi++)
            #pragma unroll
            for (int ni = 0; ni < 4; ni++)
                acc[mi][ni] = __builtin_amdgcn_wmma_f32_16x16x32_bf16(
                    false, a[mi].v, false, b[ni].v, (short)0, acc[mi][ni], false, false);

        if (more) storeStage(buf ^ 1);        // fill the other buffer (read side is `buf`)
        __syncthreads();
    }

    // C/D layout: VGPR g -> row (half*8 + g), lane -> col (lane%16)
    #pragma unroll
    for (int mi = 0; mi < 2; mi++) {
        #pragma unroll
        for (int ni = 0; ni < 4; ni++) {
            int col = n0 + wn + ni * 16 + r16;
            size_t rowb = m0 + wm + mi * 16 + half * 8;
            if (col < Nstore) {
                if constexpr (TRANSC) {
                    // transposed store: 8 consecutive M per lane -> one b128
                    auto pk = [&](float x, float y) -> unsigned {
                        if (RELU) { x = fmaxf(x, 0.f); y = fmaxf(y, 0.f); }
                        return f2bf(x) | (f2bf(y) << 16);
                    };
                    uint4 u;
                    u.x = pk(acc[mi][ni][0], acc[mi][ni][1]);
                    u.y = pk(acc[mi][ni][2], acc[mi][ni][3]);
                    u.z = pk(acc[mi][ni][4], acc[mi][ni][5]);
                    u.w = pk(acc[mi][ni][6], acc[mi][ni][7]);
                    *(uint4*)&Cp[(size_t)col * (size_t)ldc + rowb] = u;
                } else {
                    #pragma unroll
                    for (int g = 0; g < 8; g++) {
                        float v = acc[mi][ni][g];
                        if (RELU) v = fmaxf(v, 0.0f);
                        Cp[(rowb + g) * (size_t)ldc + col] = (unsigned short)f2bf(v);
                    }
                }
            }
        }
    }
}

// ---------------------------------------------------------------------------
// Prep / tail kernels
// ---------------------------------------------------------------------------

// objstate [8192][401] f32 -> bf16 padded [8192][416]
__global__ __launch_bounds__(256)
void k_cvt_obj(const float* __restrict__ src, unsigned short* __restrict__ dst)
{
    int idx = blockIdx.x * 256 + threadIdx.x;
    if (idx >= 8192 * 416) return;
    int i = idx / 416, j = idx - i * 416;
    dst[idx] = (j < 401) ? (unsigned short)f2bf(src[(size_t)i * 401 + j]) : (unsigned short)0;
}

// W [nOut][kIn] f32 -> Wnk [Nd][Kd] bf16 pad-copy (already K-contiguous)
__global__ __launch_bounds__(256)
void k_pad(const float* __restrict__ src, unsigned short* __restrict__ dst,
           int nOut, int kIn, int Nd, int Kd)
{
    int idx = blockIdx.x * 256 + threadIdx.x;
    if (idx >= Nd * Kd) return;
    int n = idx / Kd, k = idx - n * Kd;
    dst[idx] = (n < nOut && k < kIn) ? (unsigned short)f2bf(src[(size_t)n * kIn + k])
                                     : (unsigned short)0;
}

// x4[i] = concat(X3[i,0:5], class_onehot[i,0:5]) . W3     (X3 stored transposed [128][8192])
__global__ __launch_bounds__(256)
void k_x4(const unsigned short* __restrict__ X3t, const float* __restrict__ co,
          const float* __restrict__ W3, float* __restrict__ x4)
{
    int i = blockIdx.x * 256 + threadIdx.x;
    if (i >= 8192) return;
    float s = 0.f;
    #pragma unroll
    for (int j = 0; j < 5; j++)
        s += bf2f(X3t[(size_t)j * 8192 + i]) * W3[j] + co[(size_t)i * 5 + j] * W3[5 + j];
    x4[i] = s;
}

// gcn[row] = fm_w[row,:] . x4 + fm_b[row]
__global__ __launch_bounds__(256)
void k_gcn(const float* __restrict__ fm_w, const float* __restrict__ fm_b,
           const float* __restrict__ x4, float* __restrict__ gcn)
{
    __shared__ float red[256];
    int row = blockIdx.x;
    float s = 0.f;
    for (int c = threadIdx.x; c < 8192; c += 256)
        s += fm_w[(size_t)row * 8192 + c] * x4[c];
    red[threadIdx.x] = s;
    __syncthreads();
    for (int st = 128; st > 0; st >>= 1) {
        if (threadIdx.x < st) red[threadIdx.x] += red[threadIdx.x + st];
        __syncthreads();
    }
    if (threadIdx.x == 0) gcn[row] = red[0] + fm_b[row];
}

// fusion front-end: 1x1 conv + pool, glove/action embeds, pointwise fuse
__global__ __launch_bounds__(256)
void k_fusion(const float* __restrict__ state, const float* __restrict__ target,
              const float* __restrict__ ap,
              const float* __restrict__ c1w, const float* __restrict__ c1b,
              const float* __restrict__ egw, const float* __restrict__ egb,
              const float* __restrict__ eaw, const float* __restrict__ eab,
              const float* __restrict__ pww, const float* __restrict__ pwb,
              float* __restrict__ fusion_ws, float* __restrict__ imgfeat_ws)
{
    __shared__ float sImg[64];
    __shared__ float sCat[138];
    int tid = threadIdx.x;
    if (tid < 64) sImg[tid] = 0.f;
    __syncthreads();
    for (int t = tid; t < 64 * 49; t += 256) {
        int o = t / 49, p = t - o * 49;
        float s = c1b[o];
        const float* w = c1w + (size_t)o * 512;
        for (int c = 0; c < 512; c++) s += w[c] * state[c * 49 + p];
        atomicAdd(&sImg[o], fmaxf(s, 0.f));          // relu BEFORE pooling
    }
    __syncthreads();
    if (tid < 64) {
        float v = sImg[tid] * (1.f / 49.f);
        imgfeat_ws[tid] = v;
        sCat[tid] = v;
    } else if (tid < 128) {
        int o = tid - 64;
        float s = egb[o];
        const float* w = egw + (size_t)o * 300;
        for (int c = 0; c < 300; c++) s += w[c] * target[c];
        sCat[64 + o] = fmaxf(s, 0.f);
    } else if (tid < 138) {
        int o = tid - 128;
        float s = eab[o];
        for (int c = 0; c < 6; c++) s += eaw[o * 6 + c] * ap[c];
        sCat[128 + o] = fmaxf(s, 0.f);
    }
    __syncthreads();
    if (tid < 64) {
        float s = pwb[tid];
        const float* w = pww + (size_t)tid * 138;
        for (int c = 0; c < 138; c++) s += w[c] * sCat[c];
        fusion_ws[tid] = fmaxf(s, 0.f);
    }
}

// serial tail: hdp + LayerNorm + fa + LSTM cell + actor/critic
__global__ __launch_bounds__(1024)
void k_head(const float* __restrict__ fusion, const float* __restrict__ gcn,
            const float* __restrict__ hdp_w, const float* __restrict__ hdp_b,
            const float* __restrict__ ln_g, const float* __restrict__ ln_b,
            const float* __restrict__ fa_w, const float* __restrict__ fa_b,
            const float* __restrict__ wih, const float* __restrict__ whh,
            const float* __restrict__ bih, const float* __restrict__ bhh,
            const float* __restrict__ hx, const float* __restrict__ cx,
            const float* __restrict__ actor_w, const float* __restrict__ actor_b,
            const float* __restrict__ critic_w, const float* __restrict__ critic_b,
            const float* __restrict__ imgfeat, float* __restrict__ out)
{
    __shared__ float sOut[640], sH[512], sXl[640], sG[2048];
    __shared__ float red[1024];
    __shared__ float sMu, sVar;
    int tid = threadIdx.x;

    if (tid < 640)
        sOut[tid] = (tid < 64) ? fusion[tid] : (tid < 576 ? gcn[tid - 64] : fusion[tid - 576]);
    __syncthreads();

    if (tid < 512) {
        float s = hdp_b[tid];
        const float* w = hdp_w + (size_t)tid * 640;
        for (int c = 0; c < 640; c++) s += w[c] * sOut[c];
        sH[tid] = fmaxf(s, 0.f);
    }
    __syncthreads();

    red[tid] = (tid < 512) ? sH[tid] : 0.f;
    __syncthreads();
    for (int st = 512; st > 0; st >>= 1) {
        if (tid < st) red[tid] += red[tid + st];
        __syncthreads();
    }
    if (tid == 0) sMu = red[0] * (1.f / 512.f);
    __syncthreads();
    float mu = sMu;
    float d = (tid < 512) ? (sH[tid] - mu) : 0.f;
    red[tid] = d * d;
    __syncthreads();
    for (int st = 512; st > 0; st >>= 1) {
        if (tid < st) red[tid] += red[tid + st];
        __syncthreads();
    }
    if (tid == 0) sVar = red[0] * (1.f / 512.f);
    __syncthreads();
    float rstd = rsqrtf(sVar + 1e-5f);
    if (tid < 512) sH[tid] = ln_g[tid] * (sH[tid] - mu) * rstd + ln_b[tid];
    __syncthreads();

    if (tid < 640) {
        float s = fa_b[tid];
        const float* w = fa_w + (size_t)tid * 512;
        for (int c = 0; c < 512; c++) s += w[c] * sH[c];
        sXl[tid] = s;
    }
    __syncthreads();

    for (int g = tid; g < 2048; g += 1024) {
        float s = bih[g] + bhh[g];
        const float* w1 = wih + (size_t)g * 640;
        for (int c = 0; c < 640; c++) s += w1[c] * sXl[c];
        const float* w2 = whh + (size_t)g * 512;
        for (int c = 0; c < 512; c++) s += w2[c] * hx[c];
        sG[g] = s;
    }
    __syncthreads();

    if (tid < 512) {
        float ig = sG[tid], fg = sG[512 + tid], gg = sG[1024 + tid], og = sG[1536 + tid];
        float si = 1.f / (1.f + expf(-ig));
        float sf = 1.f / (1.f + expf(-fg));
        float so = 1.f / (1.f + expf(-og));
        float cN = sf * cx[tid] + si * tanhf(gg);
        float hN = so * tanhf(cN);
        out[519 + tid] = cN;    // cx_new
        out[7 + tid]   = hN;    // hx_new
        sH[tid] = hN;
    }
    __syncthreads();

    if (tid < 6) {
        float s = actor_b[tid];
        const float* w = actor_w + (size_t)tid * 512;
        for (int c = 0; c < 512; c++) s += w[c] * sH[c];
        out[1 + tid] = s;
    } else if (tid == 6) {
        float s = critic_b[0];
        for (int c = 0; c < 512; c++) s += critic_w[c] * sH[c];
        out[0] = s;
    } else if (tid >= 7 && tid < 71) {
        out[1031 + (tid - 7)] = imgfeat[tid - 7];
    }
}

// ---------------------------------------------------------------------------
// Launch
// ---------------------------------------------------------------------------
extern "C" void kernel_launch(void* const* d_in, const int* in_sizes, int n_in,
                              void* d_out, int out_size, void* d_ws, size_t ws_size,
                              hipStream_t stream)
{
    (void)in_sizes; (void)n_in; (void)out_size; (void)ws_size;

    const float* state    = (const float*)d_in[0];
    const float* target   = (const float*)d_in[1];
    const float* ap       = (const float*)d_in[2];
    const float* objstate = (const float*)d_in[3];
    const float* co       = (const float*)d_in[4];
    const float* hx       = (const float*)d_in[6];
    const float* cx       = (const float*)d_in[7];
    const float* c1w      = (const float*)d_in[8];
    const float* c1b      = (const float*)d_in[9];
    const float* egw      = (const float*)d_in[10];
    const float* egb      = (const float*)d_in[11];
    const float* eaw      = (const float*)d_in[12];
    const float* eab      = (const float*)d_in[13];
    const float* pww      = (const float*)d_in[14];
    const float* pwb      = (const float*)d_in[15];
    const float* Aadj     = (const float*)d_in[16];
    const float* W0       = (const float*)d_in[17];
    const float* W1       = (const float*)d_in[18];
    const float* W2       = (const float*)d_in[19];
    const float* W3       = (const float*)d_in[20];
    const float* fm_w     = (const float*)d_in[21];
    const float* fm_b     = (const float*)d_in[22];
    const float* hdp_w    = (const float*)d_in[25];
    const float* hdp_b    = (const float*)d_in[26];
    const float* ln_g     = (const float*)d_in[27];
    const float* ln_b     = (const float*)d_in[28];
    const float* fa_w     = (const float*)d_in[29];
    const float* fa_b     = (const float*)d_in[30];
    const float* wih      = (const float*)d_in[31];
    const float* whh      = (const float*)d_in[32];
    const float* bih      = (const float*)d_in[33];
    const float* bhh      = (const float*)d_in[34];
    const float* actw     = (const float*)d_in[35];
    const float* actb     = (const float*)d_in[36];
    const float* crw      = (const float*)d_in[37];
    const float* crb      = (const float*)d_in[38];
    float* out = (float*)d_out;

    // workspace carve-up (256B aligned)
    char*  base = (char*)d_ws;
    size_t off  = 0;
    auto alloc = [&](size_t bytes) -> void* {
        off = (off + 255) & ~(size_t)255;
        void* p = base + off;
        off += bytes;
        return p;
    };
    unsigned short* objB = (unsigned short*)alloc((size_t)8192 * 416 * 2);
    unsigned short* W0nk = (unsigned short*)alloc((size_t)512 * 416 * 2);
    unsigned short* W1nk = (unsigned short*)alloc((size_t)512 * 416 * 2);
    unsigned short* W2nk = (unsigned short*)alloc((size_t)128 * 416 * 2);
    unsigned short* X1   = (unsigned short*)alloc((size_t)8192 * 416 * 2);  // [M][416]
    unsigned short* Y1t  = (unsigned short*)alloc((size_t)512 * 8192 * 2);  // [N][M]
    unsigned short* X2   = (unsigned short*)alloc((size_t)8192 * 416 * 2);  // [M][416]
    unsigned short* Y2t  = (unsigned short*)alloc((size_t)128 * 8192 * 2);  // [N][M]
    unsigned short* X3t  = (unsigned short*)alloc((size_t)128 * 8192 * 2);  // [N][M]
    float* x4  = (float*)alloc((size_t)8192 * 4);
    float* gcn = (float*)alloc((size_t)512 * 4);
    float* fus = (float*)alloc((size_t)64 * 4);
    float* imf = (float*)alloc((size_t)64 * 4);

    // --- prep ---
    k_cvt_obj<<<(8192 * 416 + 255) / 256, 256, 0, stream>>>(objstate, objB);
    k_pad<<<(512 * 416 + 255) / 256, 256, 0, stream>>>(W0, W0nk, 401, 401, 512, 416);
    k_pad<<<(512 * 416 + 255) / 256, 256, 0, stream>>>(W1, W1nk, 401, 401, 512, 416);
    k_pad<<<(128 * 416 + 255) / 256, 256, 0, stream>>>(W2, W2nk, 5, 401, 128, 416);
    k_fusion<<<1, 256, 0, stream>>>(state, target, ap, c1w, c1b, egw, egb,
                                    eaw, eab, pww, pwb, fus, imf);

    // --- GCN chain via WMMA (all Kact % 32 == 0; Bt operands are [N][K]) ---
    // X1[m][n] = relu(sum_k obj[m][k] W0[n][k])          = relu(objstate @ W0^T)
    gemm_bf16_wmma<false, true,  false><<<dim3(4, 64), 256, 0, stream>>>(objB, W0nk, X1, 416, 416, 416, 416, 416);
    // Y1t[n][m] = sum_k X1[m][k] W1[n][k]                = (X1 @ W1^T)^T
    gemm_bf16_wmma<false, false, true ><<<dim3(4, 64), 256, 0, stream>>>(X1, W1nk, Y1t, 416, 416, 416, 8192, 512);
    // X2[m][n] = relu(sum_k A[m][k] Y1t[n][k])           = relu(A @ Y1)
    gemm_bf16_wmma<true,  true,  false><<<dim3(4, 64), 256, 0, stream>>>(Aadj, Y1t, X2, 8192, 8192, 8192, 416, 416);
    // Y2t[n][m] = sum_k X2[m][k] W2[n][k]                = (X2 @ W2^T)^T
    gemm_bf16_wmma<false, false, true ><<<dim3(1, 64), 256, 0, stream>>>(X2, W2nk, Y2t, 416, 416, 416, 8192, 128);
    // X3t[n][m] = relu(sum_k A[m][k] Y2t[n][k])          = (relu(A @ Y2))^T
    gemm_bf16_wmma<true,  true,  true ><<<dim3(1, 64), 256, 0, stream>>>(Aadj, Y2t, X3t, 8192, 8192, 8192, 8192, 128);

    // --- tail ---
    k_x4<<<32, 256, 0, stream>>>(X3t, co, W3, x4);
    k_gcn<<<512, 256, 0, stream>>>(fm_w, fm_b, x4, gcn);
    k_head<<<1, 1024, 0, stream>>>(fus, gcn, hdp_w, hdp_b, ln_g, ln_b, fa_w, fa_b,
                                   wih, whh, bih, bhh, hx, cx,
                                   actw, actb, crw, crb, imf, out);
}
